// GNNLayer_69655779607035
// MI455X (gfx1250) — matrix operand order repaired and checked by reading, not compiled, not run on hardware
//
#include <hip/hip_runtime.h>

typedef __attribute__((ext_vector_type(16))) __bf16 v16bf;
typedef __attribute__((ext_vector_type(8)))  float  v8f;
typedef unsigned int u32;
typedef u32 u32x4 __attribute__((ext_vector_type(4)));
typedef int  i32x4 __attribute__((ext_vector_type(4)));
typedef int  i32x8 __attribute__((ext_vector_type(8)));

#define NB 8
#define NN 2048
#define DD 128
#define LOG2E 1.44269504088896340736f

__device__ __forceinline__ v8f wmma_bf16(v16bf a, v16bf b, v8f c) {
  // D = A(16x32 bf16) x B(32x16 bf16) + C(16x16 f32)
  return __builtin_amdgcn_wmma_f32_16x16x32_bf16(false, a, false, b, (short)0, c,
                                                 false, false);
}

// Load 16 f32 (two contiguous 8-float runs) and split into hi/lo bf16 fragments.
// Element e maps to A-matrix K = khalf*16 + e (callers pass p1 = p0 + 8),
// matching the contiguous k order of the B fragments.
__device__ __forceinline__ void load16_split(const float* p0, const float* p1,
                                             v16bf& hi, v16bf& lo) {
  float4 A0 = *(const float4*)(p0);
  float4 A1 = *(const float4*)(p0 + 4);
  float4 A2 = *(const float4*)(p1);
  float4 A3 = *(const float4*)(p1 + 4);
  float av[16] = {A0.x,A0.y,A0.z,A0.w, A1.x,A1.y,A1.z,A1.w,
                  A2.x,A2.y,A2.z,A2.w, A3.x,A3.y,A3.z,A3.w};
  #pragma unroll
  for (int e = 0; e < 16; ++e) {
    float f = av[e];
    __bf16 h = (__bf16)f;
    hi[e] = h;
    lo[e] = (__bf16)(f - (float)h);
  }
}

// ---------------- k0: transpose W (128x128) and split into bf16 hi/lo --------
__global__ __launch_bounds__(256) void k0_wsplit(const float* __restrict__ W,
                                                 __bf16* __restrict__ WThi,
                                                 __bf16* __restrict__ WTlo) {
  int idx = blockIdx.x * 256 + threadIdx.x;  // 0..16383
  int k = idx >> 7;
  int c = idx & 127;
  float w = W[idx];
  __bf16 h = (__bf16)w;
  WThi[c * DD + k] = h;
  WTlo[c * DD + k] = (__bf16)(w - (float)h);
}

// ---------------- k1: h = x@W (split-bf16 WMMA, ~fp32 accurate) --------------
__global__ __launch_bounds__(128) void k1_hproj(
    const float* __restrict__ x, const __bf16* __restrict__ WThi,
    const __bf16* __restrict__ WTlo, const float* __restrict__ a,
    __bf16* __restrict__ hT, float* __restrict__ ssrc, float* __restrict__ sdst) {
  const int lane  = threadIdx.x & 31;
  const int wave  = threadIdx.x >> 5;
  const int mrow  = lane & 15;
  const int khalf = lane >> 4;
  const int rowbase = blockIdx.x * 64 + wave * 16;  // flat row over B*n
  const int row = rowbase + mrow;
  const float* xr = x + (size_t)row * DD;

  v8f acc[8] = {};
  #pragma unroll
  for (int kc = 0; kc < 4; ++kc) {
    v16bf xhi, xlo;
    load16_split(xr + kc * 32 + khalf * 16, xr + kc * 32 + khalf * 16 + 8, xhi, xlo);
    #pragma unroll
    for (int nt = 0; nt < 8; ++nt) {
      int col = nt * 16 + mrow;  // B-fragment: N = lane&15
      const __bf16* wph = WThi + col * DD + kc * 32 + khalf * 16;
      const __bf16* wpl = WTlo + col * DD + kc * 32 + khalf * 16;
      v16bf whi = *(const v16bf*)wph;
      v16bf wlo = *(const v16bf*)wpl;
      acc[nt] = wmma_bf16(xhi, whi, acc[nt]);
      acc[nt] = wmma_bf16(xhi, wlo, acc[nt]);
      acc[nt] = wmma_bf16(xlo, whi, acc[nt]);
    }
  }

  float psrc[8] = {}, pdst[8] = {};
  #pragma unroll
  for (int nt = 0; nt < 8; ++nt) {
    int col = nt * 16 + mrow;
    float as = a[col];
    float ad = a[DD + col];
    #pragma unroll
    for (int v = 0; v < 8; ++v) {
      int r  = rowbase + v + 8 * khalf;  // C layout: M = vgpr + 8*(lane>>4)
      int bb = r >> 11;
      int nr = r & (NN - 1);
      float hv = acc[nt][v];
      hT[((size_t)bb * DD + col) * NN + nr] = (__bf16)hv;
      psrc[v] += hv * as;
      pdst[v] += hv * ad;
    }
  }
  #pragma unroll
  for (int m = 1; m < 16; m <<= 1) {
    #pragma unroll
    for (int v = 0; v < 8; ++v) {
      psrc[v] += __shfl_xor(psrc[v], m, 32);
      pdst[v] += __shfl_xor(pdst[v], m, 32);
    }
  }
  if (mrow == 0) {
    #pragma unroll
    for (int v = 0; v < 8; ++v) {
      int r = rowbase + v + 8 * khalf;
      ssrc[r] = psrc[v] * LOG2E;  // pre-scale so exp == native exp2
      sdst[r] = pdst[v] * LOG2E;
    }
  }
}

// ---------------- k1b: per-batch max of s_dst --------------------------------
__global__ __launch_bounds__(256) void k1b_smax(const float* __restrict__ sdst,
                                                float* __restrict__ smax) {
  __shared__ float red[256];
  int b = blockIdx.x, t = threadIdx.x;
  float m = -3.0e38f;
  for (int j = t; j < NN; j += 256) m = fmaxf(m, sdst[b * NN + j]);
  red[t] = m;
  __syncthreads();
  for (int s = 128; s > 0; s >>= 1) {
    if (t < s) red[t] = fmaxf(red[t], red[t + s]);
    __syncthreads();
  }
  if (t == 0) smax[b] = red[0];
}

// ---------------- k2: fused  out = lrelu(softmax(lrelu(si+sj))@h + adj@h) ----
// hT tile (128 cols x 64 K) is shared by all 4 waves: one TDM tensor_load_to_lds
// per chunk, double-buffered in LDS, overlapped with the previous chunk's WMMAs.
__global__ __launch_bounds__(128) void k2_gat(
    const float* __restrict__ adj, const __bf16* __restrict__ hT,
    const float* __restrict__ ssrc, const float* __restrict__ sdst,
    const float* __restrict__ smax, float* __restrict__ out) {
  __shared__ __align__(128) __bf16 hbuf[2][DD * 64];  // 2 x 16 KB
  __shared__ float denomLds[64];

  const int lane  = threadIdx.x & 31;
  const int wave  = threadIdx.x >> 5;
  const int mrow  = lane & 15;
  const int khalf = lane >> 4;
  const int b = blockIdx.y;
  const int rowbase = blockIdx.x * 64 + wave * 16;
  const int row = rowbase + mrow;

  const float* adjRow = adj + ((size_t)b * NN + row) * NN;
  const float* sj  = sdst + (size_t)b * NN;
  const __bf16* hTb = hT + (size_t)b * DD * NN;

  const float si = ssrc[(size_t)b * NN + row];
  const float t0 = si + smax[b];
  const float m2 = fmaxf(t0, 0.2f * t0);  // exact row max (lrelu monotonic)

  // ---- TDM descriptor (wave-uniform). 2-D tensor: dim0 = n (contig), dim1 = col.
  const uint64_t gbase = (uint64_t)(uintptr_t)hTb;
  i32x8 g1;
  g1[0] = (int)(1u << 16);                 // workgroup_mask=0, data_size=2B
  g1[1] = (int)(((u32)NN & 0xFFFFu) << 16);  // tensor_dim0 = 2048 (bits 79:48)
  g1[2] = (int)(((u32)DD & 0xFFFFu) << 16);  // tensor_dim1 = 128  (bits 111:80)
  g1[3] = (int)(64u << 16);                // tile_dim0 = 64      (bits 127:112)
  g1[4] = (int)DD;                         // tile_dim1 = 128     (bits 143:128)
  g1[5] = (int)NN;                         // tensor_dim0_stride  (bits 207:160)
  g1[6] = (int)(((u32)NN & 0xFFFFu) << 16);  // tensor_dim1_stride (bits 255:208)
  g1[7] = 0;
  const u32 lds0 = (u32)(uintptr_t)&hbuf[0][0];
  const u32 lds1 = (u32)(uintptr_t)&hbuf[1][0];

  v8f accA[8] = {};  // adj @ h
  v8f accW[8] = {};  // exp-weights @ h (unnormalized)
  float dsum = 0.f;

  #define TDM_ISSUE(JJ, BUF)                                                  \
    do {                                                                      \
      if (wave == 0) {                                                        \
        uint64_t ga = gbase + (uint64_t)(JJ) * 2u;                            \
        u32x4 g0;                                                             \
        g0[0] = 1u;                           /* count=1 */                   \
        g0[1] = (BUF) ? lds1 : lds0;          /* lds_addr */                  \
        g0[2] = (u32)ga;                      /* global_addr[31:0] */         \
        g0[3] = (u32)(ga >> 32) | (2u << 30); /* addr[56:32] | type=2 */      \
        i32x4 z4 = {};                                                        \
        i32x8 z8 = {};                                                        \
        __builtin_amdgcn_tensor_load_to_lds(g0, g1, z4, z4, z8, 0);           \
      }                                                                       \
    } while (0)

  TDM_ISSUE(0, 0);
  if (wave == 0) __builtin_amdgcn_s_wait_tensorcnt(0);
  __syncthreads();

  const int NCH = NN / 64;  // 32 chunks of K=64
  for (int jc = 0; jc < NCH; ++jc) {
    const int jbase = jc * 64;
    const int buf = jc & 1;
    if (jc + 1 < NCH) TDM_ISSUE(jbase + 64, buf ^ 1);  // prefetch next tile

    #pragma unroll
    for (int sub = 0; sub < 2; ++sub) {
      const int j = jbase + sub * 32;
      v16bf ahi, alo;
      load16_split(adjRow + j + khalf * 16, adjRow + j + khalf * 16 + 8, ahi, alo);

      const float* q0 = sj + j + khalf * 16;
      const float* q1 = sj + j + khalf * 16 + 8;
      float4 S0 = *(const float4*)q0;
      float4 S1 = *(const float4*)(q0 + 4);
      float4 S2 = *(const float4*)q1;
      float4 S3 = *(const float4*)(q1 + 4);
      float sv[16] = {S0.x,S0.y,S0.z,S0.w, S1.x,S1.y,S1.z,S1.w,
                      S2.x,S2.y,S2.z,S2.w, S3.x,S3.y,S3.z,S3.w};
      v16bf wfrag;
      #pragma unroll
      for (int e = 0; e < 16; ++e) {
        float tt = si + sv[e];
        float lr = fmaxf(tt, 0.2f * tt);             // lrelu (pre-scaled log2e)
        float w  = __builtin_amdgcn_exp2f(lr - m2);  // native v_exp_f32
        dsum += w;
        wfrag[e] = (__bf16)w;
      }

      #pragma unroll
      for (int nt = 0; nt < 8; ++nt) {
        // B-frag from LDS tile: col*128B + sub*64B + khalf*32B, 32B aligned
        const __bf16* hp = &hbuf[buf][(nt * 16 + mrow) * 64 + sub * 32 + khalf * 16];
        v16bf bfrag = *(const v16bf*)hp;
        accA[nt] = wmma_bf16(ahi,   bfrag, accA[nt]);
        accA[nt] = wmma_bf16(alo,   bfrag, accA[nt]);
        accW[nt] = wmma_bf16(wfrag, bfrag, accW[nt]);
      }
    }

    if (wave == 0 && jc + 1 < NCH) __builtin_amdgcn_s_wait_tensorcnt(0);
    __syncthreads();  // next tile visible to all waves; prev buffer reusable
  }
  #undef TDM_ISSUE

  // softmax denominator: lane L holds half the row sum; partner is L^16
  float dtot = dsum + __shfl_xor(dsum, 16, 32);
  if (khalf == 0) denomLds[wave * 16 + mrow] = dtot;
  __syncthreads();

  float* op = out + ((size_t)b * NN + rowbase) * DD;
  #pragma unroll
  for (int nt = 0; nt < 8; ++nt) {
    int col = nt * 16 + mrow;
    #pragma unroll
    for (int v = 0; v < 8; ++v) {
      int r = v + 8 * khalf;
      float dn  = denomLds[wave * 16 + r];
      float val = accW[nt][v] / dn + accA[nt][v];
      op[(size_t)r * DD + col] = fmaxf(val, 0.2f * val);
    }
  }
}

// ---------------- launcher ---------------------------------------------------
extern "C" void kernel_launch(void* const* d_in, const int* in_sizes, int n_in,
                              void* d_out, int out_size, void* d_ws, size_t ws_size,
                              hipStream_t stream) {
  (void)in_sizes; (void)n_in; (void)out_size; (void)ws_size;
  const float* x   = (const float*)d_in[0];  // (8,2048,128)
  const float* adj = (const float*)d_in[1];  // (8,2048,2048)
  const float* W   = (const float*)d_in[2];  // (128,128)
  const float* a   = (const float*)d_in[3];  // (256,)
  float* out = (float*)d_out;                // (8,2048,128)

  char* ws = (char*)d_ws;
  const size_t HT_BYTES = (size_t)NB * DD * NN * 2;  // 4 MB
  __bf16* hT   = (__bf16*)ws;
  __bf16* WThi = (__bf16*)(ws + HT_BYTES);
  __bf16* WTlo = (__bf16*)(ws + HT_BYTES + 32768);
  float*  ssrc = (float*)(ws + HT_BYTES + 65536);
  float*  sdst = (float*)(ws + HT_BYTES + 65536 + (size_t)NB * NN * 4);
  float*  smax = (float*)(ws + HT_BYTES + 65536 + (size_t)2 * NB * NN * 4);

  k0_wsplit<<<(DD * DD) / 256, 256, 0, stream>>>(W, WThi, WTlo);
  k1_hproj<<<(NB * NN) / 64, 128, 0, stream>>>(x, WThi, WTlo, a, hT, ssrc, sdst);
  k1b_smax<<<NB, 256, 0, stream>>>(sdst, smax);
  k2_gat<<<dim3(NN / 64, NB), 128, 0, stream>>>(adj, hT, ssrc, sdst, smax, out);
}
